// text_experts_73607149519376
// MI455X (gfx1250) — compile-verified, matrix operand order checked by reading, not run on hardware
//
#include <hip/hip_runtime.h>
#include <stdint.h>

// Problem constants (from reference)
#define E_ 16
#define T_ 512
#define H_ 5120
#define F_ 8192

typedef __attribute__((ext_vector_type(16))) __bf16        v16bf;
typedef __attribute__((ext_vector_type(8)))  float         v8f;
typedef __attribute__((ext_vector_type(4)))  float         f32x4;
typedef __attribute__((ext_vector_type(4)))  unsigned int  u32x4;

union FragBF {
    v16bf v;
    unsigned int u[8];
};

// Native bf16 convert (RNE): backend emits v_cvt hardware ops.
__device__ __forceinline__ unsigned short f32_to_bf16(float f) {
    union { __bf16 b; unsigned short s; } c;
    c.b = (__bf16)f;
    return c.s;
}

// Padded LDS stride in 16-bit elements for a 32-wide K tile (even, 68B = 17 banks)
#define LDA 34

// ---------------------------------------------------------------------------
// Stage 1: hidden[e,t,f] = silu(x@Wg) * (x@Wu), stored bf16 into workspace.
// Block tile: M=64 tokens x N=64 f-columns (x2 for gate+up), 4 waves, K-step 32.
// ---------------------------------------------------------------------------
__global__ __launch_bounds__(128) void moe_gateup_swiglu_kernel(
    const float* __restrict__ x,          // [E,T,H]
    const float* __restrict__ gup,        // [E,H,2F]
    unsigned short* __restrict__ hid)     // [E,T,F] bf16
{
    __shared__ alignas(16) unsigned short As[64 * LDA];  // [m][k] bf16
    __shared__ alignas(16) unsigned short Bg[64 * LDA];  // transposed [n][k]
    __shared__ alignas(16) unsigned short Bu[64 * LDA];  // transposed [n][k]

    const int e     = blockIdx.z;
    const int tileM = blockIdx.y * 64;
    const int tileN = blockIdx.x * 64;
    const int tid   = threadIdx.x;
    const int lane  = tid & 31;
    const int wave  = tid >> 5;
    const int waveM = wave & 1;
    const int waveN = wave >> 1;
    const int laneM = lane & 15;
    const int kgrp  = (lane >> 4) * 8;    // 0 or 8, per WMMA 16-bit A/B layout

    const float* xe = x   + (size_t)e * T_ * H_;
    const float* we = gup + (size_t)e * H_ * (2 * F_);
    unsigned short* he = hid + (size_t)e * T_ * F_;

    v8f accG[2][2], accU[2][2];
#pragma unroll
    for (int i = 0; i < 2; ++i)
#pragma unroll
        for (int j = 0; j < 2; ++j) { v8f z = {}; accG[i][j] = z; accU[i][j] = z; }

    for (int kb = 0; kb < H_; kb += 32) {
        __syncthreads();
        // ---- stage A tile: 64 rows x 32 k, f32 -> bf16 (x is L2-resident: regular loads)
        {
            const int quad = tid & 7;      // 8 float4 per row
            const int row0 = tid >> 3;     // 16 rows per pass
#pragma unroll
            for (int p = 0; p < 4; ++p) {
                const int row = row0 + 16 * p;
                const f32x4 v = *(const f32x4*)(xe + (size_t)(tileM + row) * H_ + kb + quad * 4);
                unsigned short* d = &As[row * LDA + quad * 4];
                d[0] = f32_to_bf16(v.x); d[1] = f32_to_bf16(v.y);
                d[2] = f32_to_bf16(v.z); d[3] = f32_to_bf16(v.w);
            }
        }
        // ---- stage B tiles (gate + up): streamed once -> non-temporal loads
        {
            const int quad = tid & 15;     // 16 float4 per row
            const int row0 = tid >> 4;     // 8 rows per pass
#pragma unroll
            for (int p = 0; p < 4; ++p) {
                const int k = row0 + 8 * p;
                const float* wrow = we + (size_t)(kb + k) * (2 * F_);
                const f32x4 g = __builtin_nontemporal_load((const f32x4*)(wrow + tileN + quad * 4));
                const f32x4 u = __builtin_nontemporal_load((const f32x4*)(wrow + F_ + tileN + quad * 4));
                const int n = quad * 4;
                Bg[(n + 0) * LDA + k] = f32_to_bf16(g.x);
                Bg[(n + 1) * LDA + k] = f32_to_bf16(g.y);
                Bg[(n + 2) * LDA + k] = f32_to_bf16(g.z);
                Bg[(n + 3) * LDA + k] = f32_to_bf16(g.w);
                Bu[(n + 0) * LDA + k] = f32_to_bf16(u.x);
                Bu[(n + 1) * LDA + k] = f32_to_bf16(u.y);
                Bu[(n + 2) * LDA + k] = f32_to_bf16(u.z);
                Bu[(n + 3) * LDA + k] = f32_to_bf16(u.w);
            }
        }
        __syncthreads();

        // ---- gather WMMA fragments (packed K-pair uint loads)
        FragBF a[2], bg[2], bu[2];
#pragma unroll
        for (int i = 0; i < 2; ++i) {
            const unsigned short* ar = &As[(32 * waveM + 16 * i + laneM) * LDA];
#pragma unroll
            for (int v = 0; v < 8; ++v) {
                const int k0 = ((v < 4) ? 0 : 16) + kgrp + 2 * (v & 3);
                a[i].u[v] = *(const unsigned int*)(ar + k0);
            }
        }
#pragma unroll
        for (int j = 0; j < 2; ++j) {
            const unsigned short* gr = &Bg[(32 * waveN + 16 * j + laneM) * LDA];
            const unsigned short* ur = &Bu[(32 * waveN + 16 * j + laneM) * LDA];
#pragma unroll
            for (int v = 0; v < 8; ++v) {
                const int k0 = ((v < 4) ? 0 : 16) + kgrp + 2 * (v & 3);
                bg[j].u[v] = *(const unsigned int*)(gr + k0);
                bu[j].u[v] = *(const unsigned int*)(ur + k0);
            }
        }
#pragma unroll
        for (int i = 0; i < 2; ++i)
#pragma unroll
            for (int j = 0; j < 2; ++j) {
                accG[i][j] = __builtin_amdgcn_wmma_f32_16x16x32_bf16(
                    false, a[i].v, false, bg[j].v, (short)0, accG[i][j], false, false);
                accU[i][j] = __builtin_amdgcn_wmma_f32_16x16x32_bf16(
                    false, a[i].v, false, bu[j].v, (short)0, accU[i][j], false, false);
            }
    }

    // ---- SwiGLU epilogue + bf16 store (C/D layout: M = r + 8*(lane>=16), N = lane&15)
    const int mhi = (lane >> 4) * 8;
#pragma unroll
    for (int i = 0; i < 2; ++i)
#pragma unroll
        for (int j = 0; j < 2; ++j)
#pragma unroll
            for (int r = 0; r < 8; ++r) {
                const float g = accG[i][j][r];
                const float u = accU[i][j][r];
                const float s = __builtin_amdgcn_rcpf(1.0f + __expf(-g));
                const float h = u * g * s;
                const int m = tileM + 32 * waveM + 16 * i + r + mhi;
                const int n = tileN + 32 * waveN + 16 * j + laneM;
                he[(size_t)m * F_ + n] = f32_to_bf16(h);   // re-read by stage 2: keep in L2
            }
}

// ---------------------------------------------------------------------------
// Stage 2: out[e,t,h] = hidden(bf16) @ down_proj (f32 -> bf16 on load)
// Block tile: M=64 x N=128, 4 waves, each wave 32x64 (2x4 WMMA tiles), K-step 32.
// A tile is a pure bf16 copy -> async global->LDS (ASYNCcnt path), overlapped
// with the VGPR-converted B tile staging.
// ---------------------------------------------------------------------------
__global__ __launch_bounds__(128) void moe_down_kernel(
    const unsigned short* __restrict__ hid,  // [E,T,F] bf16
    const float* __restrict__ down,          // [E,F,H]
    float* __restrict__ out)                 // [E,T,H]
{
    __shared__ alignas(16) unsigned short As[64 * LDA];   // [m][k]
    __shared__ alignas(16) unsigned short Bt[128 * LDA];  // transposed [n][k]

    const int e     = blockIdx.z;
    const int tileM = blockIdx.y * 64;
    const int tileN = blockIdx.x * 128;
    const int tid   = threadIdx.x;
    const int lane  = tid & 31;
    const int wave  = tid >> 5;
    const int waveM = wave & 1;
    const int waveN = wave >> 1;
    const int laneM = lane & 15;
    const int kgrp  = (lane >> 4) * 8;

    const unsigned short* he = hid + (size_t)e * T_ * F_;
    const float* de = down + (size_t)e * F_ * H_;
    float* oe = out + (size_t)e * T_ * H_;

    // Per-thread A-copy geometry (fixed across k loop)
    const int arow = tid >> 1;               // 64 rows, 128 threads
    const int ah0  = (tid & 1) * 8;          // first 8-bf16 group
    const int ah1  = (ah0 ^ 8) + 16;         // second 8-bf16 group
    const unsigned alds0 = (unsigned)(uintptr_t)&As[arow * LDA + ah0];
    const unsigned alds1 = (unsigned)(uintptr_t)&As[arow * LDA + ah1];

    v8f acc[2][4];
#pragma unroll
    for (int i = 0; i < 2; ++i)
#pragma unroll
        for (int j = 0; j < 4; ++j) { v8f z = {}; acc[i][j] = z; }

    for (int kb = 0; kb < F_; kb += 32) {
        __syncthreads();
        // ---- A tile: async DMA global -> LDS (16B per lane per op, no VGPR round-trip)
        {
            const unsigned short* g0 = he + (size_t)(tileM + arow) * F_ + kb + ah0;
            const unsigned short* g1 = he + (size_t)(tileM + arow) * F_ + kb + ah1;
            asm volatile("global_load_async_to_lds_b128 %0, %1, off"
                         :: "v"(alds0), "v"(g0) : "memory");
            asm volatile("global_load_async_to_lds_b128 %0, %1, off"
                         :: "v"(alds1), "v"(g1) : "memory");
        }
        // ---- B tile: 32 k-rows x 128 cols, streamed once -> non-temporal, f32->bf16
        {
            const int quad = tid & 31;     // 32 float4 per row
            const int row0 = tid >> 5;     // 4 rows per pass
#pragma unroll
            for (int p = 0; p < 8; ++p) {
                const int k = row0 + 4 * p;
                const f32x4 b = __builtin_nontemporal_load(
                    (const f32x4*)(de + (size_t)(kb + k) * H_ + tileN + quad * 4));
                const int n = quad * 4;
                Bt[(n + 0) * LDA + k] = f32_to_bf16(b.x);
                Bt[(n + 1) * LDA + k] = f32_to_bf16(b.y);
                Bt[(n + 2) * LDA + k] = f32_to_bf16(b.z);
                Bt[(n + 3) * LDA + k] = f32_to_bf16(b.w);
            }
        }
        // Wave-local completion of async LDS writes, then block-wide visibility.
        asm volatile("s_wait_asynccnt 0" ::: "memory");
        __syncthreads();

        FragBF a[2], b[4];
#pragma unroll
        for (int i = 0; i < 2; ++i) {
            const unsigned short* ar = &As[(32 * waveM + 16 * i + laneM) * LDA];
#pragma unroll
            for (int v = 0; v < 8; ++v) {
                const int k0 = ((v < 4) ? 0 : 16) + kgrp + 2 * (v & 3);
                a[i].u[v] = *(const unsigned int*)(ar + k0);
            }
        }
#pragma unroll
        for (int j = 0; j < 4; ++j) {
            const unsigned short* br = &Bt[(64 * waveN + 16 * j + laneM) * LDA];
#pragma unroll
            for (int v = 0; v < 8; ++v) {
                const int k0 = ((v < 4) ? 0 : 16) + kgrp + 2 * (v & 3);
                b[j].u[v] = *(const unsigned int*)(br + k0);
            }
        }
#pragma unroll
        for (int i = 0; i < 2; ++i)
#pragma unroll
            for (int j = 0; j < 4; ++j)
                acc[i][j] = __builtin_amdgcn_wmma_f32_16x16x32_bf16(
                    false, a[i].v, false, b[j].v, (short)0, acc[i][j], false, false);
    }

    // ---- f32 writeback, never re-read -> non-temporal stores
    const int mhi = (lane >> 4) * 8;
#pragma unroll
    for (int i = 0; i < 2; ++i)
#pragma unroll
        for (int j = 0; j < 4; ++j)
#pragma unroll
            for (int r = 0; r < 8; ++r) {
                const int m = tileM + 32 * waveM + 16 * i + r + mhi;
                const int n = tileN + 64 * waveN + 16 * j + laneM;
                __builtin_nontemporal_store(acc[i][j][r], oe + (size_t)m * H_ + n);
            }
}

// ---------------------------------------------------------------------------
extern "C" void kernel_launch(void* const* d_in, const int* in_sizes, int n_in,
                              void* d_out, int out_size, void* d_ws, size_t ws_size,
                              hipStream_t stream) {
    (void)in_sizes; (void)n_in; (void)out_size; (void)ws_size;
    const float* x    = (const float*)d_in[0];
    const float* gup  = (const float*)d_in[1];
    const float* down = (const float*)d_in[2];
    float* out = (float*)d_out;
    unsigned short* hid = (unsigned short*)d_ws;   // [E,T,F] bf16 staging (128 MiB)

    dim3 blk(128, 1, 1);
    dim3 g1(F_ / 64, T_ / 64, E_);    // 128 x 8 x 16
    dim3 g2(H_ / 128, T_ / 64, E_);   // 40 x 8 x 16
    hipLaunchKernelGGL(moe_gateup_swiglu_kernel, g1, blk, 0, stream, x, gup, hid);
    hipLaunchKernelGGL(moe_down_kernel, g2, blk, 0, stream, hid, down, out);
}
